// GCN_9534827397796
// MI455X (gfx1250) — compile-verified, matrix operand order checked by reading, not run on hardware
//
#include <hip/hip_runtime.h>

typedef __attribute__((ext_vector_type(16))) __bf16 v16bf;
typedef __attribute__((ext_vector_type(8)))  __bf16 v8bf;
typedef __attribute__((ext_vector_type(8)))  float  v8f;

#define NN 8192   // number of nodes
#define DD 256    // feature dim (in == out)

// ---------------------------------------------------------------------------
// WMMA fragment loaders (wave32, 16x16x32 bf16).
// ISA layout (CDNA5 ISA): for 16-bit A (16x32) / B (32x16),
// lane L: row/col = L&15 ; lanes 0-15 hold K = 0-7 (regs 0-3) and 16-23
// (regs 4-7); lanes 16-31 hold K = 8-15 and 24-31.  => two contiguous
// 16-byte loads per lane from row-major [outer x K] storage.
// ---------------------------------------------------------------------------
__device__ __forceinline__ v16bf load_frag_bf16(const __bf16* __restrict__ base,
                                                int ld, int k0, int lane) {
  const int r    = lane & 15;
  const int koff = (lane >> 4) << 3;          // 0 or 8
  const __bf16* p = base + (size_t)r * ld + k0 + koff;
  v8bf lo = *(const v8bf*)(p);                // K = koff .. koff+7
  v8bf hi = *(const v8bf*)(p + 16);           // K = 16+koff .. 16+koff+7
  v16bf f;
#pragma unroll
  for (int i = 0; i < 8; ++i) { f[i] = lo[i]; f[i + 8] = hi[i]; }
  return f;
}

// Same layout, but source is fp32: load 2x32B and convert to bf16 in-register.
__device__ __forceinline__ v16bf load_frag_f32(const float* __restrict__ base,
                                               int ld, int k0, int lane) {
  const int r    = lane & 15;
  const int koff = (lane >> 4) << 3;
  const float* p = base + (size_t)r * ld + k0 + koff;
  v16bf f;
#pragma unroll
  for (int i = 0; i < 8; ++i) f[i]     = (__bf16)p[i];
#pragma unroll
  for (int i = 0; i < 8; ++i) f[i + 8] = (__bf16)p[16 + i];
  return f;
}

// ---------------------------------------------------------------------------
// Pass 1: dis[row] = rsqrt( rowsum(A) + 2 )           (streams A once, 256MB)
// ---------------------------------------------------------------------------
__global__ __launch_bounds__(256) void k_rowsum(const float* __restrict__ A,
                                                float* __restrict__ dis) {
  const int row = blockIdx.x;
  const float* p = A + (size_t)row * NN;
  float s = 0.0f;
#pragma unroll 4
  for (int i = threadIdx.x; i < NN; i += 256) s += p[i];
  __shared__ float red[256];
  red[threadIdx.x] = s;
  __syncthreads();
  for (int off = 128; off >= 32; off >>= 1) {
    if (threadIdx.x < off) red[threadIdx.x] += red[threadIdx.x + off];
    __syncthreads();
  }
  if (threadIdx.x < 32) {
    float v = red[threadIdx.x];
#pragma unroll
    for (int off = 16; off > 0; off >>= 1) v += __shfl_down(v, off, 32);
    if (threadIdx.x == 0) dis[row] = rsqrtf(v + 2.0f);
  }
}

// ---------------------------------------------------------------------------
// Pass 2: XsT[c][k] = bf16( dis[k] * X[k][c] )   (transposed scaled features)
// ---------------------------------------------------------------------------
__global__ __launch_bounds__(256) void k_make_xst(const float* __restrict__ X,
                                                  const float* __restrict__ dis,
                                                  __bf16* __restrict__ XsT) {
  const int tid = blockIdx.x * 256 + threadIdx.x;   // 2M elements
  const int c = tid >> 13;          // 0..255
  const int k = tid & (NN - 1);     // 0..8191  (writes coalesced along k)
  XsT[(size_t)c * NN + k] = (__bf16)(dis[k] * X[(size_t)k * DD + c]);
}

// ---------------------------------------------------------------------------
// Pass 3: Wbf = bf16(W)   (W is (out,in) row-major == B operand K-major)
// ---------------------------------------------------------------------------
__global__ __launch_bounds__(256) void k_make_wbf(const float* __restrict__ W,
                                                  __bf16* __restrict__ Wbf) {
  const int i = blockIdx.x * 256 + threadIdx.x;     // 65536 elements
  Wbf[i] = (__bf16)W[i];
}

// ---------------------------------------------------------------------------
// Pass 4: AXbf = bf16( diag(dis) * (A @ Xs + 2*Xs) )
//   A: 8192x8192 fp32 (converted to bf16 in-register)
//   XsT: 256x8192 bf16 (B operand, K-major per column)
//   block = 8 waves; wave = 16 rows x 128 cols (8 accumulators)
//   block covers 64 rows x 256 cols -> grid 128, A streamed exactly once.
// ---------------------------------------------------------------------------
__global__ __launch_bounds__(256) void k_gemm1(const float* __restrict__ A,
                                               const float* __restrict__ X,
                                               const float* __restrict__ dis,
                                               const __bf16* __restrict__ XsT,
                                               __bf16* __restrict__ AXbf) {
  const int wid  = threadIdx.x >> 5;
  const int lane = threadIdx.x & 31;
  const int r0 = blockIdx.x * 64 + (wid >> 1) * 16;   // 16-row tile
  const int c0 = (wid & 1) * 128;                     // 128-col half

  v8f acc[8];
#pragma unroll
  for (int t = 0; t < 8; ++t) acc[t] = (v8f){0.f, 0.f, 0.f, 0.f, 0.f, 0.f, 0.f, 0.f};

  const float* Arows = A + (size_t)r0 * NN;

  for (int k0 = 0; k0 < NN; k0 += 32) {
    v16bf a = load_frag_f32(Arows, NN, k0, lane);
#pragma unroll
    for (int t = 0; t < 8; ++t) {
      v16bf b = load_frag_bf16(XsT + (size_t)(c0 + t * 16) * NN, NN, k0, lane);
      acc[t] = __builtin_amdgcn_wmma_f32_16x16x32_bf16(
          false, a, false, b, (short)0, acc[t], false, false);
    }
  }

  // Epilogue: +2*Xs (diagonal term) and row scaling, emit bf16 for gemm2.
  // Acc layout: VGPR j, lane l -> row = r0 + j + 8*(l>>4), col = c0_t + (l&15)
  const int rbase = r0 + ((lane >> 4) << 3);
  const int nlane = lane & 15;
#pragma unroll
  for (int j = 0; j < 8; ++j) {
    const int row = rbase + j;
    const float dr = dis[row];
#pragma unroll
    for (int t = 0; t < 8; ++t) {
      const int col = c0 + t * 16 + nlane;
      float v = acc[t][j];
      v = dr * (v + 2.0f * dr * X[(size_t)row * DD + col]);
      AXbf[(size_t)row * DD + col] = (__bf16)v;
    }
  }
}

// ---------------------------------------------------------------------------
// Pass 5: out = relu( AX @ W^T + b )    (K = 256)
//   wave = 16 rows x 64 cols (4 accumulators); block = 32 rows x 256 cols.
// ---------------------------------------------------------------------------
__global__ __launch_bounds__(256) void k_gemm2(const __bf16* __restrict__ AXbf,
                                               const __bf16* __restrict__ Wbf,
                                               const float* __restrict__ bias,
                                               float* __restrict__ out) {
  const int wid  = threadIdx.x >> 5;
  const int lane = threadIdx.x & 31;
  const int r0 = blockIdx.x * 32 + (wid >> 2) * 16;
  const int c0 = (wid & 3) * 64;

  v8f acc[4];
#pragma unroll
  for (int t = 0; t < 4; ++t) acc[t] = (v8f){0.f, 0.f, 0.f, 0.f, 0.f, 0.f, 0.f, 0.f};

  const __bf16* Arows = AXbf + (size_t)r0 * DD;

#pragma unroll
  for (int k0 = 0; k0 < DD; k0 += 32) {
    v16bf a = load_frag_bf16(Arows, DD, k0, lane);
#pragma unroll
    for (int t = 0; t < 4; ++t) {
      v16bf b = load_frag_bf16(Wbf + (size_t)(c0 + t * 16) * DD, DD, k0, lane);
      acc[t] = __builtin_amdgcn_wmma_f32_16x16x32_bf16(
          false, a, false, b, (short)0, acc[t], false, false);
    }
  }

  const int rbase = r0 + ((lane >> 4) << 3);
  const int nlane = lane & 15;
#pragma unroll
  for (int t = 0; t < 4; ++t) {
    const int col = c0 + t * 16 + nlane;
    const float bv = bias[col];
#pragma unroll
    for (int j = 0; j < 8; ++j) {
      const int row = rbase + j;
      float v = acc[t][j] + bv;
      out[(size_t)row * DD + col] = v > 0.0f ? v : 0.0f;
    }
  }
}

// ---------------------------------------------------------------------------
// Host-side launcher. Inputs (setup_inputs order): X, A, W, b. Output fp32.
// Workspace layout: dis (32KB) | XsT (4MB) | AXbf (4MB) | Wbf (128KB)
// ---------------------------------------------------------------------------
extern "C" void kernel_launch(void* const* d_in, const int* in_sizes, int n_in,
                              void* d_out, int out_size, void* d_ws, size_t ws_size,
                              hipStream_t stream) {
  const float* X = (const float*)d_in[0];
  const float* A = (const float*)d_in[1];
  const float* W = (const float*)d_in[2];
  const float* b = (const float*)d_in[3];
  float* out = (float*)d_out;

  char* ws = (char*)d_ws;
  float*  dis  = (float*)ws;                                   // 8192 * 4B
  __bf16* XsT  = (__bf16*)(ws + 32768);                        // 256*8192*2B
  __bf16* AXbf = (__bf16*)(ws + 32768 + 4194304);              // 8192*256*2B
  __bf16* Wbf  = (__bf16*)(ws + 32768 + 2 * 4194304);          // 256*256*2B

  k_rowsum  <<<NN,              256, 0, stream>>>(A, dis);
  k_make_xst<<<(NN * DD) / 256, 256, 0, stream>>>(X, dis, XsT);
  k_make_wbf<<<(DD * DD) / 256, 256, 0, stream>>>(W, Wbf);
  k_gemm1   <<<NN / 64,         256, 0, stream>>>(A, X, dis, XsT, AXbf);
  k_gemm2   <<<NN / 32,         256, 0, stream>>>(AXbf, Wbf, b, out);
}
